// JointEncoder_87789131530737
// MI455X (gfx1250) — matrix operand order repaired and checked by reading, not compiled, not run on hardware
//
#include <hip/hip_runtime.h>

// ---- problem constants (from reference) ----
constexpr int   B_  = 2,  N_ = 6,  C_ = 192, HF = 64, WF = 176;
constexpr int   HR  = 32, WR = 1024;
constexpr int   K1  = 768;                      // hidden width
constexpr int   NPTS = B_ * HR * WR;            // 65536 points
constexpr int   NPROJ = B_ * N_ * HR * WR;      // 393216 projections
constexpr float H_IMGF = 256.0f, W_IMGF = 704.0f;
constexpr int   OUT_RANGE_ELEMS = B_ * (C_ + 1) * HR * WR;   // 12,648,448
constexpr int   IMG_ELEMS       = B_ * N_ * C_ * HF * WF;    // 25,952,256

// LDS padding: 776 halfs/row => 1552B => 388 dwords, 388 % 64 = 4 (staggers banks)
constexpr int   HPAD = 776;

typedef __attribute__((ext_vector_type(16))) _Float16 v16h;
typedef __attribute__((ext_vector_type(8)))  _Float16 v8h;
typedef __attribute__((ext_vector_type(8)))  float    v8f;

// ------------------------------------------------------------------
// 1) W2 (768x192 f32, row-major) -> W2^T (192x768 f16) so each WMMA B
//    fragment is two contiguous 16B loads per lane.
// ------------------------------------------------------------------
__global__ void k_prep_w2t(const float* __restrict__ w2, _Float16* __restrict__ w2t) {
    int j = blockIdx.x * 256 + threadIdx.x;
    if (j >= K1 * C_) return;
    int k = j / C_;
    int n = j - k * C_;
    w2t[n * K1 + k] = (_Float16)w2[k * C_ + n];
}

// ------------------------------------------------------------------
// 2) Geometry: one thread per (b,n,h,w). Writes range_cam_coors and the
//    packed gather index (r*WF+c, or -1 when invalid) to scratch.
// ------------------------------------------------------------------
__global__ void k_geom(const float* __restrict__ pts, const float* __restrict__ rimg,
                       const float* __restrict__ l2i,
                       float* __restrict__ coors, int* __restrict__ nidx) {
    int i = blockIdx.x * 256 + threadIdx.x;
    if (i >= NPROJ) return;
    int w = i % WR;
    int h = (i / WR) % HR;
    int n = (i / (WR * HR)) % N_;
    int b = i / (WR * HR * N_);
    int p = (b * HR + h) * WR + w;

    float px = pts[p * 3 + 0], py = pts[p * 3 + 1], pz = pts[p * 3 + 2];
    const float* M = l2i + (size_t)(b * N_ + n) * 16;
    float cx = fmaf(M[0], px, fmaf(M[1],  py, fmaf(M[2],  pz, M[3])));
    float cy = fmaf(M[4], px, fmaf(M[5],  py, fmaf(M[6],  pz, M[7])));
    float cz = fmaf(M[8], px, fmaf(M[9],  py, fmaf(M[10], pz, M[11])));
    float d = fmaxf(cz, 1e-5f);
    float x = cx / d, y = cy / d;

    bool no_point = !(px == 0.0f && py == 0.0f && pz == 0.0f);
    bool in_img   = (x >= 0.0f) && (x < W_IMGF) && (y >= 0.0f) && (y < H_IMGF);
    bool valid    = no_point && in_img;

    coors[(size_t)i * 2 + 0] = valid ? y : -1.0f;
    coors[(size_t)i * 2 + 1] = valid ? x : -1.0f;

    bool valid2 = valid && (rimg[p] > 0.0f);
    int r = (int)(y * (float)HF / H_IMGF); r = r < 0 ? 0 : (r > HF - 1 ? HF - 1 : r);
    int c = (int)(x * (float)WF / W_IMGF); c = c < 0 ? 0 : (c > WF - 1 ? WF - 1 : c);
    nidx[i] = valid2 ? (r * WF + c) : -1;
}

// ------------------------------------------------------------------
// 3) Gather+sum over cameras. Block = one point, thread = one channel.
//    img_features (104MB) is L2-resident (192MB L2); prefetch the 6
//    scattered gather addresses before the dependent loads.
// ------------------------------------------------------------------
__global__ void __launch_bounds__(C_) k_gather(const float* __restrict__ img,
                                               const float* __restrict__ rimg,
                                               const int* __restrict__ nidx,
                                               float* __restrict__ out_range) {
    int pi = blockIdx.x;                 // 0 .. NPTS-1
    int ch = threadIdx.x;                // 0 .. 191
    int w = pi % WR;
    int h = (pi / WR) % HR;
    int b = pi / (WR * HR);

    __shared__ int sidx[N_];
    if (ch < N_) sidx[ch] = nidx[((b * N_ + ch) * HR + h) * WR + w];
    __syncthreads();

    const float* ap[N_];
#pragma unroll
    for (int n = 0; n < N_; ++n) {
        int idx = sidx[n];
        ap[n] = (idx >= 0) ? &img[(size_t)((b * N_ + n) * C_ + ch) * (HF * WF) + idx] : nullptr;
        if (ap[n]) __builtin_prefetch(ap[n], 0, 0);   // global_prefetch_b8
    }
    float s = 0.0f;
#pragma unroll
    for (int n = 0; n < N_; ++n)
        if (ap[n]) s += *ap[n];

    size_t base = (size_t)b * (C_ + 1) * HR * WR;
    out_range[base + (size_t)(ch + 1) * HR * WR + h * WR + w] = s;
    if (ch == 0)
        out_range[base + (size_t)h * WR + w] = rimg[(b * HR + h) * WR + w];
}

// ------------------------------------------------------------------
// 4) MLP via WMMA. Block: 32 rows x full 192 cols = 2 Mtiles x 12 Ntiles
//    = 24 wave-tiles on 8 waves (3 each). Hidden (K=3 -> 768, relu) is
//    computed in VALU, stored to LDS as f16. GEMM: K=768 in 24 steps of
//    v_wmma_f32_16x16x32_f16, f32 accumulate, bias epilogue.
// ------------------------------------------------------------------
__global__ void __launch_bounds__(256) k_mlp(const float* __restrict__ pts,
                                             const float* __restrict__ w1,
                                             const float* __restrict__ b1,
                                             const _Float16* __restrict__ w2t,
                                             const float* __restrict__ b2,
                                             float* __restrict__ pos) {
    __shared__ _Float16 sh[32 * HPAD];   // 49,664 B
    __shared__ float    sp[32 * 3];

    const int m0  = blockIdx.x * 32;
    const int tid = threadIdx.x;

    if (tid < 96) sp[tid] = pts[(size_t)m0 * 3 + tid];
    __syncthreads();

    // hidden layer: h[row][k] = relu(p . w1[:,k] + b1[k]), cast to f16
    for (int e = tid; e < 32 * K1; e += 256) {
        int row = e / K1;
        int k   = e - row * K1;
        float hv = fmaf(sp[row * 3 + 0], w1[k],
                   fmaf(sp[row * 3 + 1], w1[K1 + k],
                   fmaf(sp[row * 3 + 2], w1[2 * K1 + k], b1[k])));
        sh[row * HPAD + k] = (_Float16)fmaxf(hv, 0.0f);
    }
    __syncthreads();

    const int wave = tid >> 5;
    const int lane = tid & 31;
    const int lrow = lane & 15;          // row within tile (A) / col within tile (B,D)
    const int kg   = lane >> 4;          // K-group half of the wave

    for (int t = wave; t < 24; t += 8) {
        int mt = t / 12;
        int nt = t - mt * 12;
        int n  = nt * 16 + lrow;                      // output column
        const _Float16* bp = w2t + (size_t)n * K1;

        v8f acc = {};
        for (int ks = 0; ks < 24; ++ks) {
            int k0 = ks * 32;
            // A fragment (16x32 f16): lane holds K = kg*8 + {0..7} and kg*8+16 + {0..7}
            const _Float16* ap = &sh[(mt * 16 + lrow) * HPAD + k0 + kg * 8];
            v8h a0 = *(const v8h*)ap;
            v8h a1 = *(const v8h*)(ap + 16);
            // B fragment (32x16 f16): lane holds K = kg*8 + {0..7} and kg*8+16 + {0..7}
            const _Float16* bq = bp + k0 + kg * 8;
            v8h q0 = *(const v8h*)bq;
            v8h q1 = *(const v8h*)(bq + 16);
            v16h A, Bf;
#pragma unroll
            for (int i = 0; i < 8; ++i) {
                A[i] = a0[i];  A[i + 8] = a1[i];
                Bf[i] = q0[i]; Bf[i + 8] = q1[i];
            }
            acc = __builtin_amdgcn_wmma_f32_16x16x32_f16(
                false, A, false, Bf, (short)0, acc, false, false);
        }

        float bias = b2[n];
#pragma unroll
        for (int r = 0; r < 8; ++r) {
            int gm = m0 + mt * 16 + kg * 8 + r;       // D layout: lanes16-31 hold M+8
            pos[(size_t)gm * C_ + n] = acc[r] + bias;
        }
    }
}

// ------------------------------------------------------------------
// 5) img_out = img_features (float4 streaming copy)
// ------------------------------------------------------------------
__global__ void k_copy(const float4* __restrict__ src, float4* __restrict__ dst, int n4) {
    for (int i = blockIdx.x * blockDim.x + threadIdx.x; i < n4; i += gridDim.x * blockDim.x)
        dst[i] = src[i];
}

// ------------------------------------------------------------------
// 6) Scatter-add pos into img_out. Block = one (b,n,h,w), thread = channel.
// ------------------------------------------------------------------
__global__ void __launch_bounds__(C_) k_scatter(const int* __restrict__ nidx,
                                                const float* __restrict__ pos,
                                                float* __restrict__ img_out) {
    int bi  = blockIdx.x;                // 0 .. NPROJ-1
    int idx = nidx[bi];
    if (idx < 0) return;
    int w = bi % WR;
    int h = (bi / WR) % HR;
    int n = (bi / (WR * HR)) % N_;
    int b = bi / (WR * HR * N_);
    int ch = threadIdx.x;
    float v = pos[(size_t)((b * HR + h) * WR + w) * C_ + ch];
    atomicAdd(&img_out[(size_t)((b * N_ + n) * C_ + ch) * (HF * WF) + idx], v);
}

// ------------------------------------------------------------------
extern "C" void kernel_launch(void* const* d_in, const int* in_sizes, int n_in,
                              void* d_out, int out_size, void* d_ws, size_t ws_size,
                              hipStream_t stream) {
    (void)in_sizes; (void)n_in; (void)out_size; (void)ws_size;
    const float* img  = (const float*)d_in[0];
    const float* rimg = (const float*)d_in[1];
    const float* pts  = (const float*)d_in[2];
    const float* l2i  = (const float*)d_in[3];
    const float* w1   = (const float*)d_in[4];
    const float* b1   = (const float*)d_in[5];
    const float* w2   = (const float*)d_in[6];
    const float* b2   = (const float*)d_in[7];

    float* out       = (float*)d_out;
    float* out_range = out;                       // (B, 1+C, HR, WR)
    float* img_out   = out + OUT_RANGE_ELEMS;     // (B, N, C, HF, WF)
    float* coors     = img_out + IMG_ELEMS;       // (B, N, HR, WR, 2)

    char* ws = (char*)d_ws;
    _Float16* w2t = (_Float16*)ws;                               // 294,912 B
    int*      nidx = (int*)(ws + 294912);                        // 1,572,864 B
    float*    pos  = (float*)(ws + 294912 + 1572864);            // 48 MB, 16B aligned

    k_prep_w2t<<<(K1 * C_ + 255) / 256, 256, 0, stream>>>(w2, w2t);
    k_geom<<<(NPROJ + 255) / 256, 256, 0, stream>>>(pts, rimg, l2i, coors, nidx);
    k_gather<<<NPTS, C_, 0, stream>>>(img, rimg, nidx, out_range);
    k_mlp<<<NPTS / 32, 256, 0, stream>>>(pts, w1, b1, w2t, b2, pos);
    k_copy<<<4096, 256, 0, stream>>>((const float4*)img, (float4*)img_out, IMG_ELEMS / 4);
    k_scatter<<<NPROJ, C_, 0, stream>>>(nidx, pos, img_out);
}
